// Model_83554293777064
// MI455X (gfx1250) — matrix operand order repaired and checked by reading, not compiled, not run on hardware
//
#include <hip/hip_runtime.h>
#include <math.h>

typedef __attribute__((ext_vector_type(16))) _Float16 v16h;
typedef __attribute__((ext_vector_type(8)))  float    v8f;

#define NNODES 50000
#define NEDGES 800000
#define NNON   400000
#define NGRAPH 512
#define EMB    64

// ======================= utility kernels =======================
__global__ void fill_f32(float* p, float v, int n) {
  int t = blockIdx.x * blockDim.x + threadIdx.x;
  if (t < n) p[t] = v;
}
__global__ void copy_f32(float* __restrict__ d, const float* __restrict__ s, int n) {
  int t = blockIdx.x * blockDim.x + threadIdx.x;
  if (t < n) d[t] = s[t];
}

__device__ __forceinline__ float atomAddF(float* p, float v) {
  return unsafeAtomicAdd(p, v);   // global_atomic_add_f32
}
__device__ __forceinline__ void atomMaxF(float* addr, float val) {
  if (val >= 0.0f) atomicMax((int*)addr, __float_as_int(val));
  else             atomicMin((unsigned int*)addr, __float_as_uint(val));
}

// Pack f32 weight (K x N row-major) into WMMA-B f16 per-lane layout.
// B fragment for V_WMMA_F32_16X16X32_F16: lane l holds col n=l&15,
// K = (l>>4)*8 + h (h<8), 16 + (l>>4)*8 + (h-8) (h>=8) within each 32-row K block.
// half index = ((ntile*kblks + kblk)*32 + lane)*16 + h.
__global__ void pack_weight(const float* __restrict__ W, _Float16* __restrict__ out,
                            int K, int N, int kblks, int ntiles) {
  int t = blockIdx.x * blockDim.x + threadIdx.x;
  int total = kblks * ntiles * 512;
  if (t >= total) return;
  int h    = t & 15;
  int lane = (t >> 4) & 31;
  int blk  = t >> 9;
  int kblk  = blk % kblks;
  int ntile = blk / kblks;
  int row = kblk * 32 + (h >> 3) * 16 + (lane >> 4) * 8 + (h & 7);
  int col = ntile * 16 + (lane & 15);
  float v = (row < K && col < N) ? W[row * N + col] : 0.0f;
  out[t] = (_Float16)v;
}

// ======================= WMMA GEMM =======================
// C[M x Nout] = act( fetch(A)[M x K] @ Wpacked[K x Np] + bias (+ Cadd) )
// MODE: 0 = direct, 1 = direct + A2 (elementwise add), 2 = gather2 (A[i0]+A[i1])
// KK in {16,64}. LDCC: compile-time ldc (0 = runtime). NFULL: Nout == 16*ntiles.
template <bool AHALF, bool CHALF, int KK, int MODE, bool CADD, bool ACT,
          bool BIAS, int LDCC, bool NFULL, bool FULL>
__device__ __forceinline__ void gemm_body(
    const void* __restrict__ A_, const float* __restrict__ A2,
    const int* __restrict__ idx0, const int* __restrict__ idx1, int idxStride, int lda,
    const _Float16* __restrict__ Wp, const float* __restrict__ bias,
    void* __restrict__ C_, const float* __restrict__ Cadd, int ldcDyn,
    int M, int Nout, int ntiles, int r0, int lane) {
  constexpr int NSEG = KK / 16;          // 1 (K=16) or 4 (K=64)
  constexpr int KB   = (KK + 31) / 32;   // K blocks of 32
  const int ldc = LDCC ? LDCC : ldcDyn;
  int m = lane & 15, hi = lane >> 4, koff = hi * 8;
  int row = r0 + m;

  union { v16h v; _Float16 h[16]; } a0, a1;
#pragma unroll
  for (int i = 0; i < 16; ++i) { a0.h[i] = (_Float16)0.0f; a1.h[i] = (_Float16)0.0f; }

  if (FULL || row < M) {
    float seg[NSEG][8];
    if (AHALF) {
      const _Float16* ph = (const _Float16*)A_ + (size_t)row * lda;
#pragma unroll
      for (int s = 0; s < NSEG; ++s)
#pragma unroll
        for (int h = 0; h < 8; ++h) seg[s][h] = (float)ph[s * 16 + koff + h];
    } else {
      const float* p0;
      const float* p1 = nullptr;
      if (MODE == 2) {
        int ra = idx0[(size_t)row * idxStride];
        int rb = idx1[(size_t)row * idxStride];
        p0 = (const float*)A_ + (size_t)ra * lda;
        p1 = (const float*)A_ + (size_t)rb * lda;
      } else {
        p0 = (const float*)A_ + (size_t)row * lda;
        if (MODE == 1) p1 = A2 + (size_t)row * lda;
      }
#pragma unroll
      for (int s = 0; s < NSEG; ++s) {
        const float4* q0 = (const float4*)(p0 + s * 16 + koff);
        float4 v0 = q0[0], v1 = q0[1];
        if (MODE != 0) {
          const float4* q1 = (const float4*)(p1 + s * 16 + koff);
          float4 w0 = q1[0], w1 = q1[1];
          v0.x += w0.x; v0.y += w0.y; v0.z += w0.z; v0.w += w0.w;
          v1.x += w1.x; v1.y += w1.y; v1.z += w1.z; v1.w += w1.w;
        }
        seg[s][0] = v0.x; seg[s][1] = v0.y; seg[s][2] = v0.z; seg[s][3] = v0.w;
        seg[s][4] = v1.x; seg[s][5] = v1.y; seg[s][6] = v1.z; seg[s][7] = v1.w;
      }
    }
#pragma unroll
    for (int h = 0; h < 8; ++h) {
      a0.h[h] = (_Float16)seg[0][h];
      if (NSEG == 4) {
        a0.h[8 + h] = (_Float16)seg[1][h];
        a1.h[h]     = (_Float16)seg[2][h];
        a1.h[8 + h] = (_Float16)seg[3][h];
      }
    }
  }

  float*    Cf = (float*)C_;
  _Float16* Ch = (_Float16*)C_;
  const v16h* wp = (const v16h*)Wp;
  const int mr0 = r0 + hi * 8;          // first of this lane's 8 rows
  for (int t = 0; t < ntiles; ++t) {
    v16h b0 = wp[(t * KB + 0) * 32 + lane];
    int col = t * 16 + m;
    float cb = 0.0f;
    if (BIAS) cb = (NFULL || col < Nout) ? bias[col] : 0.0f;
    v8f c = { cb, cb, cb, cb, cb, cb, cb, cb };
    c = __builtin_amdgcn_wmma_f32_16x16x32_f16(false, a0.v, false, b0, (short)0, c, false, false);
    if (KB == 2) {
      v16h b1 = wp[(t * KB + 1) * 32 + lane];
      c = __builtin_amdgcn_wmma_f32_16x16x32_f16(false, a1.v, false, b1, (short)0, c, false, false);
    }
    if (NFULL || col < Nout) {
      float*       cp = Cf + (size_t)mr0 * ldc + col;
      _Float16*    hp = Ch + (size_t)mr0 * ldc + col;
      const float* ap = CADD ? (Cadd + (size_t)mr0 * ldc + col) : nullptr;
#pragma unroll
      for (int r = 0; r < 8; ++r) {
        if (FULL || (mr0 + r) < M) {
          float val = c[r];
          if (CADD) val += ap[(size_t)r * ldc];
          if (ACT)  val = val > 0.0f ? val : 0.01f * val;
          if (CHALF) hp[(size_t)r * ldc] = (_Float16)val;
          else       cp[(size_t)r * ldc] = val;
        }
      }
    }
  }
}

template <bool AHALF, bool CHALF, int KK, int MODE, bool CADD, bool ACT,
          bool BIAS, int LDCC, bool NFULL>
__global__ void gemm_wmma(const void* __restrict__ A_, const float* __restrict__ A2,
                          const int* __restrict__ idx0, const int* __restrict__ idx1,
                          int idxStride, int lda,
                          const _Float16* __restrict__ Wp, const float* __restrict__ bias,
                          void* __restrict__ C_, const float* __restrict__ Cadd, int ldcDyn,
                          int M, int Nout, int ntiles) {
  int lane = threadIdx.x & 31;
  int wv   = threadIdx.x >> 5;
  int r0   = (blockIdx.x * (int)(blockDim.x >> 5) + wv) << 4;
  if (r0 >= M) return;                 // wave-uniform: EXEC all-ones for WMMA
  if (r0 + 16 <= M)
    gemm_body<AHALF, CHALF, KK, MODE, CADD, ACT, BIAS, LDCC, NFULL, true>(
        A_, A2, idx0, idx1, idxStride, lda, Wp, bias, C_, Cadd, ldcDyn, M, Nout, ntiles, r0, lane);
  else
    gemm_body<AHALF, CHALF, KK, MODE, CADD, ACT, BIAS, LDCC, NFULL, false>(
        A_, A2, idx0, idx1, idxStride, lda, Wp, bias, C_, Cadd, ldcDyn, M, Nout, ntiles, r0, lane);
}

// ======================= per-edge / segment kernels =======================
__global__ void gen_msg(const float* __restrict__ o, const _Float16* __restrict__ e,
                        const int* __restrict__ src, const int* __restrict__ dst,
                        float* __restrict__ agg, int total) {
  int t = blockIdx.x * blockDim.x + threadIdx.x;
  if (t >= total) return;
  int j = t >> 6, kf = t & 63;
  int s = src[j], d = dst[j];
  float m = o[(size_t)s * EMB + kf] + (float)e[(size_t)j * EMB + kf];
  m = (m > 0.0f ? m : 0.0f) + 1e-7f;
  atomAddF(&agg[(size_t)d * EMB + kf], m);
}

__global__ void trans_score(const float* __restrict__ q, const float* __restrict__ kk,
                            const _Float16* __restrict__ ee,
                            const int* __restrict__ src, const int* __restrict__ dst,
                            float* __restrict__ score, float* __restrict__ smax, int nE) {
  int j = blockIdx.x * blockDim.x + threadIdx.x;
  if (j >= nE) return;
  int s = src[j], d = dst[j];
  const float4* qp = (const float4*)(q + (size_t)d * EMB);
  const float4* kp = (const float4*)(kk + (size_t)s * EMB);
  const _Float16* ep = ee + (size_t)j * EMB;
  float acc = 0.0f;
#pragma unroll
  for (int i = 0; i < 16; ++i) {
    float4 qv = qp[i], kv = kp[i];
    acc += qv.x * (kv.x + (float)ep[4 * i + 0]);
    acc += qv.y * (kv.y + (float)ep[4 * i + 1]);
    acc += qv.z * (kv.z + (float)ep[4 * i + 2]);
    acc += qv.w * (kv.w + (float)ep[4 * i + 3]);
  }
  acc *= 0.125f;
  score[j] = acc;
  atomMaxF(&smax[d], acc);
}

__global__ void trans_expsum(const float* __restrict__ score, const float* __restrict__ smax,
                             const int* __restrict__ dst, float* __restrict__ aexp,
                             float* __restrict__ denom, int nE) {
  int j = blockIdx.x * blockDim.x + threadIdx.x;
  if (j >= nE) return;
  int d = dst[j];
  float sm = smax[d];
  if (isinf(sm)) sm = 0.0f;
  float a = expf(score[j] - sm);
  aexp[j] = a;
  atomAddF(&denom[d], a);
}

__global__ void trans_msg(const float* __restrict__ v, const _Float16* __restrict__ ee,
                          const float* __restrict__ aexp, const float* __restrict__ denom,
                          const int* __restrict__ src, const int* __restrict__ dst,
                          float* __restrict__ attn, int total) {
  int t = blockIdx.x * blockDim.x + threadIdx.x;
  if (t >= total) return;
  int j = t >> 6, kf = t & 63;
  int s = src[j], d = dst[j];
  float al = aexp[j] / (denom[d] + 1e-16f);
  float val = (v[(size_t)s * EMB + kf] + (float)ee[(size_t)j * EMB + kf]) * al;
  atomAddF(&attn[(size_t)d * EMB + kf], val);
}

__global__ void pool_add(const float* __restrict__ o, const int* __restrict__ batch,
                         float* __restrict__ glob, float* __restrict__ cnt, int total) {
  int t = blockIdx.x * blockDim.x + threadIdx.x;
  if (t >= total) return;
  int n = t >> 6, kf = t & 63;
  int b = batch[n];
  atomAddF(&glob[(size_t)b * EMB + kf], o[t]);
  if (kf == 0) atomAddF(&cnt[b], 1.0f);
}
__global__ void pool_div(float* __restrict__ glob, const float* __restrict__ cnt, int total) {
  int t = blockIdx.x * blockDim.x + threadIdx.x;
  if (t >= total) return;
  glob[t] /= fmaxf(cnt[t >> 6], 1.0f);
}

// ======================= host side =======================
struct ParamPtrs {
  const float *x2hW, *x2hB, *e2hW, *e2hB;
  const float* genW[6];
  const float *Wq[6], *bq[6], *Wk[6], *bk[6], *Wv[6], *bv[6], *We[6], *Ws[6], *bs[6];
  const float *hW1[5], *hb1[5], *hW2[5], *hb2[5];  // stop, add_node, node_attr, add_edge, edge_attr
};

static inline int ceil_div(long a, int b) { return (int)((a + b - 1) / b); }

// ---- wrappers over template instantiations ----
// full-width (ldc=64, Nout=64) layer GEMMs: branch-free epilogue
static void gemm_p64(hipStream_t s, const float* A, const _Float16* Wp, const float* bias,
                     float* C, int M) {
  gemm_wmma<false, false, 64, 0, false, false, true, 64, true><<<ceil_div(M, 128), 256, 0, s>>>(
      A, nullptr, nullptr, nullptr, 0, 64, Wp, bias, C, nullptr, 64, M, 64, 4);
}
static void gemm_act(hipStream_t s, const float* A, const _Float16* Wp, const float* bias,
                     float* C, int M) {
  gemm_wmma<false, false, 64, 0, false, true, true, 64, true><<<ceil_div(M, 128), 256, 0, s>>>(
      A, nullptr, nullptr, nullptr, 0, 64, Wp, bias, C, nullptr, 64, M, 64, 4);
}
static void gemm_gather_act(hipStream_t s, const float* A, const int* i0, const int* i1,
                            int istride, const _Float16* Wp, const float* bias, float* C, int M) {
  gemm_wmma<false, false, 64, 2, false, true, true, 64, true><<<ceil_div(M, 128), 256, 0, s>>>(
      A, nullptr, i0, i1, istride, 64, Wp, bias, C, nullptr, 64, M, 64, 4);
}
static void gemm_gen(hipStream_t s, const float* A, const float* A2, const _Float16* Wp,
                     float* C, const float* Cadd, int M) {
  gemm_wmma<false, false, 64, 1, true, false, false, 64, true><<<ceil_div(M, 128), 256, 0, s>>>(
      A, A2, nullptr, nullptr, 0, 64, Wp, nullptr, C, Cadd, 64, M, 64, 4);
}
static void gemm_ws(hipStream_t s, const float* A, const _Float16* Wp, const float* bias,
                    float* C, const float* Cadd, int M) {
  gemm_wmma<false, false, 64, 0, true, false, true, 64, true><<<ceil_div(M, 128), 256, 0, s>>>(
      A, nullptr, nullptr, nullptr, 0, 64, Wp, bias, C, Cadd, 64, M, 64, 4);
}
static void gemm_e2h(hipStream_t s, const float* A, const _Float16* Wp, const float* bias,
                     _Float16* C, int M) {
  gemm_wmma<false, true, 16, 0, false, false, true, 64, true><<<ceil_div(M, 128), 256, 0, s>>>(
      A, nullptr, nullptr, nullptr, 0, 16, Wp, bias, C, nullptr, 64, M, 64, 4);
}
static void gemm_ee(hipStream_t s, const _Float16* A, const _Float16* Wp, _Float16* C, int M) {
  gemm_wmma<true, true, 64, 0, false, false, false, 64, true><<<ceil_div(M, 128), 256, 0, s>>>(
      A, nullptr, nullptr, nullptr, 0, 64, Wp, nullptr, C, nullptr, 64, M, 64, 4);
}
// head output GEMMs: dynamic ldc / ragged Nout
static void gemm_head(hipStream_t s, const float* A, const _Float16* Wp, const float* bias,
                      float* C, int ldc, int M, int Nout, int ntiles) {
  gemm_wmma<false, false, 64, 0, false, false, true, 0, false><<<ceil_div(M, 128), 256, 0, s>>>(
      A, nullptr, nullptr, nullptr, 0, 64, Wp, bias, C, nullptr, ldc, M, Nout, ntiles);
}

extern "C" void kernel_launch(void* const* d_in, const int* in_sizes, int n_in,
                              void* d_out, int out_size, void* d_ws, size_t ws_size,
                              hipStream_t stream) {
  (void)n_in; (void)out_size; (void)ws_size;
  auto F = [&](int i) { return (const float*)d_in[i]; };

  // --- detect pytree flattening scheme via sizes ---
  int scheme;
  if (in_sizes[0] == NNODES) scheme = 2;            // fully sorted (batch first)
  else if (in_sizes[5] == 1024) scheme = 1;         // top insertion, params jax-sorted
  else scheme = 0;                                  // full insertion order
  const float* x;  const float* edge_attr;  const int* ei;  const int* nei;  const int* batch;
  if (scheme == 2) {
    batch = (const int*)d_in[0]; edge_attr = F(1); ei = (const int*)d_in[2];
    nei = (const int*)d_in[3]; x = F(88);
  } else {
    x = F(0); edge_attr = F(1); ei = (const int*)d_in[2];
    nei = (const int*)d_in[3]; batch = (const int*)d_in[4];
  }
  ParamPtrs P;
  if (scheme == 0) {
    P.x2hW = F(5); P.x2hB = F(6); P.e2hW = F(7); P.e2hB = F(8);
    for (int l = 0; l < 6; ++l) P.genW[l] = F(9 + l);
    for (int l = 0; l < 6; ++l) {
      int t = 15 + 9 * l;
      P.Wq[l]=F(t); P.bq[l]=F(t+1); P.Wk[l]=F(t+2); P.bk[l]=F(t+3); P.Wv[l]=F(t+4);
      P.bv[l]=F(t+5); P.We[l]=F(t+6); P.Ws[l]=F(t+7); P.bs[l]=F(t+8);
    }
    for (int h = 0; h < 5; ++h) {
      int t = 69 + 4 * h;
      P.hW1[h]=F(t); P.hb1[h]=F(t+1); P.hW2[h]=F(t+2); P.hb2[h]=F(t+3);
    }
  } else {
    int b = (scheme == 1) ? 5 : 4;
    P.e2hW = F(b); P.e2hB = F(b + 1);
    for (int l = 0; l < 6; ++l) P.genW[l] = F(b + 2 + l);
    int sortpos[5] = {4, 1, 3, 0, 2};  // canonical -> sorted chunk (heads sorted alphabetically)
    for (int h = 0; h < 5; ++h) {
      int t = b + 8 + 4 * sortpos[h];
      P.hW1[h]=F(t); P.hW2[h]=F(t+1); P.hb1[h]=F(t+2); P.hb2[h]=F(t+3);
    }
    for (int l = 0; l < 6; ++l) {
      int t = b + 28 + 9 * l;  // We,Wk,Wq,Ws,Wv,bk,bq,bs,bv
      P.We[l]=F(t); P.Wk[l]=F(t+1); P.Wq[l]=F(t+2); P.Ws[l]=F(t+3); P.Wv[l]=F(t+4);
      P.bk[l]=F(t+5); P.bq[l]=F(t+6); P.bs[l]=F(t+7); P.bv[l]=F(t+8);
    }
    P.x2hW = F(b + 82); P.x2hB = F(b + 83);
  }

  // --- workspace layout ---
  char* wptr = (char*)d_ws;
  auto alloc = [&](size_t bytes) { void* p = wptr; wptr += (bytes + 255) & ~(size_t)255; return p; };
  float* o    = (float*)alloc((size_t)NNODES * EMB * 4);
  float* agg  = (float*)alloc((size_t)NNODES * EMB * 4);
  float* qb   = (float*)alloc((size_t)NNODES * EMB * 4);
  float* kb   = (float*)alloc((size_t)NNODES * EMB * 4);
  float* vb   = (float*)alloc((size_t)NNODES * EMB * 4);
  float* attn = (float*)alloc((size_t)NNODES * EMB * 4);
  _Float16* e  = (_Float16*)alloc((size_t)NEDGES * EMB * 2);
  _Float16* ee = (_Float16*)alloc((size_t)NEDGES * EMB * 2);
  float* score = (float*)alloc((size_t)NEDGES * 4);
  float* aexp  = (float*)alloc((size_t)NEDGES * 4);
  float* smax  = (float*)alloc((size_t)NNODES * 4);
  float* denom = (float*)alloc((size_t)NNODES * 4);
  float* glob  = (float*)alloc((size_t)NGRAPH * EMB * 4);
  float* cnt   = (float*)alloc((size_t)NGRAPH * 4);
  _Float16* pwpool = (_Float16*)alloc(262144 * 2);
  float* hid = (float*)ee;   // reuse ee region after layers: 400000*64 f32 fits exactly

  // --- pack weights ---
  _Float16* pcur = pwpool;
  auto packW = [&](const float* W, int K, int N, int kblks, int ntiles) {
    _Float16* dst = pcur;
    int total = kblks * ntiles * 512;
    pack_weight<<<ceil_div(total, 256), 256, 0, stream>>>(W, dst, K, N, kblks, ntiles);
    pcur += total;
    return dst;
  };
  _Float16* Px2h = packW(P.x2hW, 64, 64, 2, 4);
  _Float16* Pe2h = packW(P.e2hW, 16, 64, 1, 4);   // K=16: single 32-K block (zero padded)
  _Float16 *Pgen[6], *Pq[6], *Pk[6], *Pv[6], *Pwe[6], *Pws[6];
  for (int l = 0; l < 6; ++l) {
    Pgen[l] = packW(P.genW[l], 64, 64, 2, 4);
    Pq[l]  = packW(P.Wq[l], 64, 64, 2, 4);
    Pk[l]  = packW(P.Wk[l], 64, 64, 2, 4);
    Pv[l]  = packW(P.Wv[l], 64, 64, 2, 4);
    Pwe[l] = packW(P.We[l], 64, 64, 2, 4);
    Pws[l] = packW(P.Ws[l], 64, 64, 2, 4);
  }
  int headN[5]  = {1, 72, 32, 1, 16};
  int headNT[5] = {1, 5, 2, 1, 1};
  _Float16 *PW1[5], *PW2[5];
  for (int h = 0; h < 5; ++h) {
    PW1[h] = packW(P.hW1[h], 64, 64, 2, 4);
    PW2[h] = packW(P.hW2[h], 64, headN[h], 2, headNT[h]);
  }

  const int* src = ei;
  const int* dst = ei + NEDGES;
  const int* ne0 = nei;
  const int* ne1 = nei + NNON;
  const int nodeTot = NNODES * EMB;
  const int edgeTot = NEDGES * EMB;

  // --- embeddings ---
  gemm_p64(stream, x, Px2h, P.x2hB, o, NNODES);
  gemm_e2h(stream, edge_attr, Pe2h, P.e2hB, e, NEDGES);

  // --- 6 layers ---
  for (int l = 0; l < 6; ++l) {
    // gen conv: o = o + (segsum(relu(o[src]+e)+eps) + o) @ W
    fill_f32<<<ceil_div(nodeTot, 256), 256, 0, stream>>>(agg, 0.0f, nodeTot);
    gen_msg<<<ceil_div(edgeTot, 256), 256, 0, stream>>>(o, e, src, dst, agg, edgeTot);
    gemm_gen(stream, agg, o, Pgen[l], o, o, NNODES);
    // trans conv
    gemm_p64(stream, o, Pq[l], P.bq[l], qb, NNODES);
    gemm_p64(stream, o, Pk[l], P.bk[l], kb, NNODES);
    gemm_p64(stream, o, Pv[l], P.bv[l], vb, NNODES);
    gemm_ee(stream, e, Pwe[l], ee, NEDGES);
    fill_f32<<<ceil_div(NNODES, 256), 256, 0, stream>>>(smax, -INFINITY, NNODES);
    fill_f32<<<ceil_div(NNODES, 256), 256, 0, stream>>>(denom, 0.0f, NNODES);
    trans_score<<<ceil_div(NEDGES, 256), 256, 0, stream>>>(qb, kb, ee, src, dst, score, smax, NEDGES);
    trans_expsum<<<ceil_div(NEDGES, 256), 256, 0, stream>>>(score, smax, dst, aexp, denom, NEDGES);
    copy_f32<<<ceil_div(nodeTot, 256), 256, 0, stream>>>(attn, o, nodeTot);  // attn = o + msgs
    trans_msg<<<ceil_div(edgeTot, 256), 256, 0, stream>>>(vb, ee, aexp, denom, src, dst, attn, edgeTot);
    gemm_ws(stream, o, Pws[l], P.bs[l], o, attn, NNODES);   // o = attn + o@Ws + bs
  }

  // --- global mean pool ---
  fill_f32<<<ceil_div(NGRAPH * EMB, 256), 256, 0, stream>>>(glob, 0.0f, NGRAPH * EMB);
  fill_f32<<<ceil_div(NGRAPH, 256), 256, 0, stream>>>(cnt, 0.0f, NGRAPH);
  pool_add<<<ceil_div(nodeTot, 256), 256, 0, stream>>>(o, batch, glob, cnt, nodeTot);
  pool_div<<<ceil_div(NGRAPH * EMB, 256), 256, 0, stream>>>(glob, cnt, NGRAPH * EMB);

  // --- heads (outputs concatenated in return order) ---
  float* out = (float*)d_out;
  float* outStop     = out;
  float* outAddNode  = outStop + NGRAPH * 1;
  float* outNodeAttr = outAddNode + (size_t)NNODES * 72;
  float* outAddEdge  = outNodeAttr + (size_t)NNODES * 32;
  float* outEdgeAttr = outAddEdge + (size_t)NNON * 1;

  gemm_act(stream, glob, PW1[0], P.hb1[0], hid, NGRAPH);
  gemm_head(stream, hid, PW2[0], P.hb2[0], outStop, 1, NGRAPH, 1, 1);

  gemm_act(stream, o, PW1[1], P.hb1[1], hid, NNODES);
  gemm_head(stream, hid, PW2[1], P.hb2[1], outAddNode, 72, NNODES, 72, 5);

  gemm_act(stream, o, PW1[2], P.hb1[2], hid, NNODES);
  gemm_head(stream, hid, PW2[2], P.hb2[2], outNodeAttr, 32, NNODES, 32, 2);

  gemm_gather_act(stream, o, ne0, ne1, 1, PW1[3], P.hb1[3], hid, NNON);
  gemm_head(stream, hid, PW2[3], P.hb2[3], outAddEdge, 1, NNON, 1, 1);

  gemm_gather_act(stream, o, src, dst, 2, PW1[4], P.hb1[4], hid, NEDGES / 2);
  gemm_head(stream, hid, PW2[4], P.hb2[4], outEdgeAttr, 16, NEDGES / 2, 16, 1);
}
